// HierarchicalPooling_2843268350301
// MI455X (gfx1250) — matrix operand order repaired and checked by reading, not compiled, not run on hardware
//
#include <hip/hip_runtime.h>
#include <hip/hip_bf16.h>
#include <math.h>

typedef _Float16 v16h __attribute__((ext_vector_type(16)));
typedef _Float16 v8h  __attribute__((ext_vector_type(8)));
typedef _Float16 v4h  __attribute__((ext_vector_type(4)));
typedef float    v8f  __attribute__((ext_vector_type(8)));
typedef float    v4f  __attribute__((ext_vector_type(4)));

#define N_NODES 131072
#define C_CH    64
#define K_SUP   32
#define B_GR    16
#define TN      128        // nodes per workgroup tile
#define EPSF    1e-9f

// ---- output layout (floats, tuple concatenated flat) ----
#define OUT_POOL_OFF 0                                   // [B,K,C] = 32768
#define OUT_S_OFF    (B_GR * K_SUP * C_CH)               // 32768
#define OUT_SC_OFF   (OUT_S_OFF + N_NODES * K_SUP)       // 4227072 (6 scalars)
#define OUT_MU_OFF   (OUT_SC_OFF + 6)                    // 4227078 ([B,K,2] = 1024)

// ---- workspace layout (floats) ----
#define WS_SUMS 0        // sum_s        [B][K]     512
#define WS_POS  512      // sum_s_pos    [B][K][2] 1024
#define WS_PQ   1536     // s^T |pos|^2  [K]         32
#define WS_ENT  1568     // entropy partial sum       1
#define WS_TOTAL 1600

// ---- LDS arena offsets (bytes) ----
#define A_R0   0        // 16896: xh f16 [128][64]  -> later logits/s f32 [128][33]
#define A_R1   16896    // 16384: xT f16 [64][128]  (pooling B operand)
#define A_R2   33280    // 8192 : w1t f16 [64][64]  -> later sT16 f16 [32][128]
#define A_R3   41472    // 4096 : w2t f16 [32][64]
#define A_R4   45568    // 16384: hh  f16 [128][64]
#define A_B1   61952    // 256  : b1 f32[64]
#define A_B2   62208    // 128  : b2 f32[32]
#define A_POS  62336    // 1024 : pos f32[128][2]
#define A_BAT  63360    // 512  : batch int[128]
#define A_RED  63872    // 1024 : reduction f32[256]
#define A_SLO  64896    // 64   : segLo int[16]
#define A_SHI  64960    // 64   : segHi int[16]
#define A_SIZE 65024

__device__ __forceinline__ v16h make_v16(v8h lo, v8h hi) {
    v16h r;
#pragma unroll
    for (int i = 0; i < 8; ++i) { r[i] = lo[i]; r[i + 8] = hi[i]; }
    return r;
}

// A fragment: source row-major with contiguous K (stride-64 rows).
// element e -> K = kk + e%8 + (e/8)*16 + lh*8   (ISA 16-bit A 16x32 layout)
__device__ __forceinline__ v16h load_frag_A(const _Float16* p /* base+row*64+kk */, int lh) {
    const _Float16* q = p + lh * 8;
    return make_v16(*(const v8h*)q, *(const v8h*)(q + 16));
}

// B fragment: source transposed so 16 K values are contiguous.
// element e -> K = kk + e%8 + (e/8)*16 + lh*8   (ISA 16-bit B K x 16 layout)
__device__ __forceinline__ v16h load_frag_B(const _Float16* p /* wT+col*64+kk */, int lh) {
    const _Float16* q = p + lh * 8;
    return make_v16(*(const v8h*)q, *(const v8h*)(q + 16));
}

__device__ __forceinline__ v8f wmma_f16(v16h a, v16h b, v8f c) {
    return __builtin_amdgcn_wmma_f32_16x16x32_f16(false, a, false, b, (short)0, c, false, false);
}

__global__ __launch_bounds__(256) void hp_main_kernel(
    const float* __restrict__ x, const int* __restrict__ batch,
    const float* __restrict__ pos, const float* __restrict__ u,
    const float* __restrict__ W1, const float* __restrict__ b1,
    const float* __restrict__ W2, const float* __restrict__ b2,
    const float* __restrict__ scaling, const float* __restrict__ amask,
    float* __restrict__ out_pool, float* __restrict__ out_s,
    float* __restrict__ ws)
{
    __shared__ __align__(16) unsigned char arena[A_SIZE];

    _Float16* xh   = (_Float16*)(arena + A_R0);
    float*    lgtF = (float*)   (arena + A_R0);   // aliases xh after GEMM1
    _Float16* xT   = (_Float16*)(arena + A_R1);
    _Float16* w1t  = (_Float16*)(arena + A_R2);
    _Float16* sT16 = (_Float16*)(arena + A_R2);   // aliases w1t after GEMM1
    _Float16* w2t  = (_Float16*)(arena + A_R3);
    _Float16* hh   = (_Float16*)(arena + A_R4);
    float*    b1f  = (float*)   (arena + A_B1);
    float*    b2f  = (float*)   (arena + A_B2);
    float*    posL = (float*)   (arena + A_POS);
    int*      batL = (int*)     (arena + A_BAT);
    float*    red  = (float*)   (arena + A_RED);
    int*      segLo= (int*)     (arena + A_SLO);
    int*      segHi= (int*)     (arena + A_SHI);

    const int tid  = threadIdx.x;
    const int w    = tid >> 5;        // wave id 0..7
    const int lane = tid & 31;
    const int lh   = lane >> 4;       // half-wave select
    const int lm   = lane & 15;
    const int n0   = blockIdx.x * TN;

    // Warm L2/L0 with this tile's gumbel rows; not consumed until after both
    // GEMMs, so latency is fully hidden (global_prefetch_b8).
    if (tid < TN) __builtin_prefetch(u + (size_t)(n0 + tid) * K_SUP, 0, 1);

    // ---------------- phase 0: stage operands into LDS ----------------
    for (int i = tid; i < C_CH * C_CH; i += 256) {       // W1^T (out-major)
        int cin = i >> 6, cout = i & 63;
        w1t[cout * 64 + cin] = (_Float16)W1[i];
    }
    for (int i = tid; i < C_CH * K_SUP; i += 256) {      // W2^T (out-major)
        int cin = i >> 5, cout = i & 31;
        w2t[cout * 64 + cin] = (_Float16)W2[i];
    }
    if (tid < C_CH)  b1f[tid] = b1[tid];
    if (tid < K_SUP) b2f[tid] = b2[tid];
    if (tid < 16) { segLo[tid] = 0x7fffffff; segHi[tid] = -1; }
    if (tid < TN) {
        batL[tid] = batch[n0 + tid];
        posL[tid * 2]     = pos[(size_t)(n0 + tid) * 2];
        posL[tid * 2 + 1] = pos[(size_t)(n0 + tid) * 2 + 1];
    }
    // x tile: streaming NT B128 loads, dual layout (row-major f16 + transposed)
    for (int i4 = tid; i4 < TN * C_CH / 4; i4 += 256) {
        int i = i4 * 4;
        int n = i >> 6, c = i & 63;
        v4f v = __builtin_nontemporal_load((const v4f*)(x + (size_t)(n0 + n) * C_CH + c));
        v4h hv;
#pragma unroll
        for (int j = 0; j < 4; ++j) hv[j] = (_Float16)v[j];
        *(v4h*)(xh + n * 64 + c) = hv;                    // one ds_store_b64
#pragma unroll
        for (int j = 0; j < 4; ++j) xT[(c + j) * TN + n] = hv[j];
    }
    __syncthreads();
    if (tid < TN) {                                       // per-batch segment ranges
        int b = batL[tid];
        atomicMin(&segLo[b], tid);
        atomicMax(&segHi[b], tid + 1);
    }

    // ---------------- GEMM1: H = relu(X @ W1 + b1), [128x64] ----------------
#pragma unroll
    for (int nt = 0; nt < 4; ++nt) {
        float bias = b1f[nt * 16 + lm];
        v8f acc;
#pragma unroll
        for (int r = 0; r < 8; ++r) acc[r] = bias;
#pragma unroll
        for (int kk = 0; kk < 64; kk += 32) {
            v16h a = load_frag_A(xh + (w * 16 + lm) * 64 + kk, lh);
            v16h b = load_frag_B(w1t + (nt * 16 + lm) * 64 + kk, lh);
            acc = wmma_f16(a, b, acc);
        }
#pragma unroll
        for (int r = 0; r < 8; ++r) {
            float v = acc[r] > 0.f ? acc[r] : 0.f;        // ReLU
            hh[(w * 16 + r + lh * 8) * 64 + nt * 16 + lm] = (_Float16)v;
        }
    }
    __syncthreads();

    // ---------------- GEMM2: L = H @ W2 + b2, scale, mask -> lgtF [128][33] ----------------
    const float scal = scaling[0];
#pragma unroll
    for (int nt = 0; nt < 2; ++nt) {
        const int k = nt * 16 + lm;
        float bias = b2f[k];
        v8f acc;
#pragma unroll
        for (int r = 0; r < 8; ++r) acc[r] = bias;
#pragma unroll
        for (int kk = 0; kk < 64; kk += 32) {
            v16h a = load_frag_A(hh + (w * 16 + lm) * 64 + kk, lh);
            v16h b = load_frag_B(w2t + k * 64 + kk, lh);
            acc = wmma_f16(a, b, acc);
        }
        float am = amask[k];
#pragma unroll
        for (int r = 0; r < 8; ++r) {
            float v = acc[r] * scal;
            if (am == 0.f) v = -1e9f;
            lgtF[(w * 16 + r + lh * 8) * 33 + k] = v;
        }
    }
    __syncthreads();

    // ---------------- gumbel-softmax per node (threads 0..127), entropy partial ----------------
    float entPart = 0.f;
    if (tid < TN) {
        const v4f* up4 = (const v4f*)(u + (size_t)(n0 + tid) * K_SUP);
        float vals[K_SUP];
        float mx = -1e30f;
#pragma unroll
        for (int k4 = 0; k4 < K_SUP / 4; ++k4) {
            v4f uu = __builtin_nontemporal_load(up4 + k4);
#pragma unroll
            for (int j = 0; j < 4; ++j) {
                int k = k4 * 4 + j;
                float g = -__logf(-__logf(uu[j] + EPSF) + EPSF);
                float v = lgtF[tid * 33 + k] + g;         // TAU == 1
                vals[k] = v;
                mx = fmaxf(mx, v);
            }
        }
        float sum = 0.f;
#pragma unroll
        for (int k = 0; k < K_SUP; ++k) { float e = __expf(vals[k] - mx); vals[k] = e; sum += e; }
        float inv = 1.f / sum;
#pragma unroll
        for (int k = 0; k < K_SUP; ++k) {
            float s = vals[k] * inv;
            lgtF[tid * 33 + k] = s;                       // s now lives in lgtF
            entPart += s * __logf(s + EPSF);
        }
    }
    red[tid] = entPart;
    __syncthreads();
#pragma unroll
    for (int off = 128; off > 0; off >>= 1) {
        if (tid < off) red[tid] += red[tid + off];
        __syncthreads();
    }
    if (tid == 0) atomicAdd(ws + WS_ENT, red[0]);

    // ---------------- write s (NT B128, coalesced) + build s^T in f16 ----------------
    for (int i4 = tid; i4 < TN * K_SUP / 4; i4 += 256) {
        int i = i4 * 4;
        int n = i >> 5, k = i & 31;
        v4f sv;
#pragma unroll
        for (int j = 0; j < 4; ++j) sv[j] = lgtF[n * 33 + k + j];
        __builtin_nontemporal_store(sv, (v4f*)(out_s + (size_t)n0 * K_SUP + i));
#pragma unroll
        for (int j = 0; j < 4; ++j) sT16[(k + j) * TN + n] = (_Float16)sv[j];
    }
    __syncthreads();

    const int bFirst = batL[0];
    const int bLast  = batL[TN - 1];

    // ---------------- segment sums (wave 0): sum_s, sum_s*pos, sum_s*|pos|^2 ----------------
    if (w == 0) {
        const int k = lane;
        for (int b = bFirst; b <= bLast; ++b) {
            int lo = segLo[b], hi = segHi[b];
            if (hi <= lo) continue;
            float as = 0.f, ax = 0.f, ay = 0.f, aq = 0.f;
            for (int n = lo; n < hi; ++n) {
                float s  = lgtF[n * 33 + k];
                float px = posL[n * 2], py = posL[n * 2 + 1];
                as += s; ax += s * px; ay += s * py; aq += s * (px * px + py * py);
            }
            atomicAdd(ws + WS_SUMS + b * K_SUP + k, as);
            atomicAdd(ws + WS_POS + (b * K_SUP + k) * 2,     ax);
            atomicAdd(ws + WS_POS + (b * K_SUP + k) * 2 + 1, ay);
            atomicAdd(ws + WS_PQ + k, aq);
        }
    }

    // ---------------- pooling GEMM: out[b] += s^T @ x (masked per segment) ----------------
    const int kt = w & 1;        // 2 k-tiles
    const int ct = w >> 1;       // 4 c-tiles
    for (int b = bFirst; b <= bLast; ++b) {
        int lo = segLo[b], hi = segHi[b];
        if (hi <= lo) continue;
        v8f acc;
#pragma unroll
        for (int r = 0; r < 8; ++r) acc[r] = 0.f;
#pragma unroll
        for (int ch = 0; ch < TN; ch += 32) {
            // A = s^T tile: row = k index, element e -> node = ch + e%8 + (e/8)*16 + lh*8
            const _Float16* ap = sT16 + (kt * 16 + lm) * TN + ch + lh * 8;
            v8h alo = *(const v8h*)ap;
            v8h ahi = *(const v8h*)(ap + 16);
            int na0 = ch + lh * 8, na1 = ch + 16 + lh * 8;
            v16h a;
#pragma unroll
            for (int i = 0; i < 8; ++i) {
                int n1 = na0 + i, n2 = na1 + i;
                a[i]     = (n1 >= lo && n1 < hi) ? alo[i] : (_Float16)0.f;
                a[i + 8] = (n2 >= lo && n2 < hi) ? ahi[i] : (_Float16)0.f;
            }
            // B = x tile: col = c, element e -> node = ch + e%8 + (e/8)*16 + lh*8 (from xT)
            const _Float16* bp = xT + (ct * 16 + lm) * TN + ch + lh * 8;
            v16h bb = make_v16(*(const v8h*)bp, *(const v8h*)(bp + 16));
            acc = wmma_f16(a, bb, acc);
        }
        float* op = out_pool + ((size_t)b * K_SUP + kt * 16) * C_CH + ct * 16 + lm;
#pragma unroll
        for (int r = 0; r < 8; ++r)
            atomicAdd(op + (size_t)(r + lh * 8) * C_CH, acc[r]);
    }
}

__global__ __launch_bounds__(256) void hp_finalize_kernel(
    const float* __restrict__ ws, const float* __restrict__ amask,
    float* __restrict__ d_out)
{
    __shared__ float muL[B_GR * K_SUP * 2];
    __shared__ float red[256];
    __shared__ float kvals[3 * K_SUP];   // [0:32) diversity, [32:64) pruning, [64:96) var

    const int tid = threadIdx.x;
    float* out_mu = d_out + OUT_MU_OFF;

    // mu = sum_s_pos / (sum_s + EPS)
    for (int i = tid; i < B_GR * K_SUP; i += 256) {
        float ss = ws[WS_SUMS + i] + EPSF;
        float mx = ws[WS_POS + i * 2]     / ss;
        float my = ws[WS_POS + i * 2 + 1] / ss;
        muL[i * 2] = mx;  muL[i * 2 + 1] = my;
        out_mu[i * 2] = mx; out_mu[i * 2 + 1] = my;
    }

    // per-k global stats
    if (tid < K_SUP) {
        float cs = 0.f, sx = 0.f, sy = 0.f;
        for (int b = 0; b < B_GR; ++b) {
            cs += ws[WS_SUMS + b * K_SUP + tid];
            sx += ws[WS_POS + (b * K_SUP + tid) * 2];
            sy += ws[WS_POS + (b * K_SUP + tid) * 2 + 1];
        }
        float avg = cs / (float)N_NODES;
        float am  = amask[tid];
        kvals[tid]          = avg * __logf(avg + EPSF);         // diversity term
        kvals[K_SUP + tid]  = fabsf(avg * (1.f - am));          // pruning term
        float ssum = cs + EPSF;
        float mgx = sx / ssum, mgy = sy / ssum;
        float epq = ws[WS_PQ + tid] / ssum;
        kvals[2 * K_SUP + tid] = epq - (mgx * mgx + mgy * mgy); // var term
    }
    __syncthreads();

    // separation: sum over b, i!=j of 1/(d^2+1)
    float acc = 0.f;
    for (int t = tid; t < B_GR * K_SUP * K_SUP; t += 256) {
        int b = t >> 10, rem = t & 1023, i = rem >> 5, j = rem & 31;
        if (i != j) {
            float dx = muL[(b * K_SUP + i) * 2]     - muL[(b * K_SUP + j) * 2];
            float dy = muL[(b * K_SUP + i) * 2 + 1] - muL[(b * K_SUP + j) * 2 + 1];
            acc += 1.f / (dx * dx + dy * dy + 1.f);
        }
    }
    red[tid] = acc;
    __syncthreads();
#pragma unroll
    for (int off = 128; off > 0; off >>= 1) {
        if (tid < off) red[tid] += red[tid + off];
        __syncthreads();
    }

    if (tid == 0) {
        float div = 0.f, prun = 0.f, var = 0.f, spars = 0.f;
        for (int k = 0; k < K_SUP; ++k) {
            div   += kvals[k];
            prun  += kvals[K_SUP + k];
            var   += kvals[2 * K_SUP + k];
            spars += amask[k];
        }
        float* sc = d_out + OUT_SC_OFF;
        sc[0] = -ws[WS_ENT] / (float)N_NODES;                       // entropy
        sc[1] = div;                                                // diversity
        sc[2] = var / (float)K_SUP;                                 // spatial
        sc[3] = prun / (float)K_SUP;                                // pruning
        sc[4] = spars / (float)K_SUP;                               // sparsity
        sc[5] = red[0] / ((float)(K_SUP * (K_SUP - 1)) + EPSF);     // separation
    }
}

extern "C" void kernel_launch(void* const* d_in, const int* in_sizes, int n_in,
                              void* d_out, int out_size, void* d_ws, size_t ws_size,
                              hipStream_t stream) {
    const float* x       = (const float*)d_in[0];
    const int*   batch   = (const int*)  d_in[1];
    const float* pos     = (const float*)d_in[2];
    const float* u       = (const float*)d_in[3];
    const float* W1      = (const float*)d_in[4];
    const float* b1      = (const float*)d_in[5];
    const float* W2      = (const float*)d_in[6];
    const float* b2      = (const float*)d_in[7];
    const float* scaling = (const float*)d_in[8];
    const float* amask   = (const float*)d_in[9];

    float* out = (float*)d_out;
    float* ws  = (float*)d_ws;

    // zero the accumulated regions (graph-capture-safe memset nodes)
    hipMemsetAsync(out, 0, (size_t)(B_GR * K_SUP * C_CH) * sizeof(float), stream);
    hipMemsetAsync(ws, 0, (size_t)WS_TOTAL * sizeof(float), stream);

    hp_main_kernel<<<N_NODES / TN, 256, 0, stream>>>(
        x, batch, pos, u, W1, b1, W2, b2, scaling, amask,
        out + OUT_POOL_OFF, out + OUT_S_OFF, ws);

    hp_finalize_kernel<<<1, 256, 0, stream>>>(ws, amask, out);
}
